// ContextualViewModel_68968584839670
// MI455X (gfx1250) — compile-verified, matrix operand order checked by reading, not run on hardware
//
#include <hip/hip_runtime.h>

// Problem constants (from reference)
#define HGRID 64
#define WGRID 64
#define NCELL (HGRID * WGRID)   // 4096
#define CTXD  512
#define HIDD  2048
#define KNN   8
#define BATCH 8
#define MTOT  (BATCH * NCELL)   // 32768

// GEMM tiling
#define BM 128
#define BN 128
#define BK 32
#define AS 40   // LDS row stride (halfs): 32 data + 4-DWORD pad (TDM pad feature)
#define BS 40   // LDS row stride (halfs) for B^T tile
#define LDS_BYTES ((BM * AS + BN * BS) * 2)   // 20480

typedef _Float16 h4   __attribute__((ext_vector_type(4)));
typedef _Float16 v8h  __attribute__((ext_vector_type(8)));
typedef _Float16 v16h __attribute__((ext_vector_type(16)));
typedef float    v8f  __attribute__((ext_vector_type(8)));
typedef unsigned int u32x4 __attribute__((ext_vector_type(4)));
typedef int          i32x4 __attribute__((ext_vector_type(4)));
typedef int          i32x8 __attribute__((ext_vector_type(8)));

// ---------------------------------------------------------------------------
// Kernel 1: convert W (f32, CTXD x HIDD) -> f16
// ---------------------------------------------------------------------------
__global__ __launch_bounds__(256) void convw_kernel(const float* __restrict__ W,
                                                    _Float16* __restrict__ WH) {
    size_t i = ((size_t)blockIdx.x * 256 + threadIdx.x) * 4;
    float4 v = *(const float4*)(W + i);
    h4 h;
    h[0] = (_Float16)v.x;
    h[1] = (_Float16)v.y;
    h[2] = (_Float16)v.z;
    h[3] = (_Float16)v.w;
    *(h4*)(WH + i) = h;
}

// ---------------------------------------------------------------------------
// Kernel 2: agg[b,n,c] = sum_k sim[n,k] * x[b, nn_idx[n,k], c]  (f32 math, f16 out)
// ---------------------------------------------------------------------------
__global__ __launch_bounds__(128) void agg_kernel(const float* __restrict__ x,
                                                  const int* __restrict__ nn_idx,
                                                  const float* __restrict__ sim,
                                                  _Float16* __restrict__ aggH) {
    __shared__ int   s_idx[KNN];
    __shared__ float s_sim[KNN];
    const int n = blockIdx.x;
    const int b = blockIdx.y;
    if (threadIdx.x < KNN) {
        s_idx[threadIdx.x] = nn_idx[n * KNN + threadIdx.x];
        s_sim[threadIdx.x] = sim[n * KNN + threadIdx.x];
    }
    __syncthreads();

    float ax = 0.f, ay = 0.f, az = 0.f, aw = 0.f;
#pragma unroll
    for (int k = 0; k < KNN; ++k) {
        const float4* row = (const float4*)(x + ((size_t)b * NCELL + s_idx[k]) * CTXD);
        float4 v = row[threadIdx.x];
        float  s = s_sim[k];
        ax += s * v.x; ay += s * v.y; az += s * v.z; aw += s * v.w;
    }
    h4 h;
    h[0] = (_Float16)ax; h[1] = (_Float16)ay; h[2] = (_Float16)az; h[3] = (_Float16)aw;
    *(h4*)(aggH + ((size_t)b * NCELL + n) * CTXD + (size_t)threadIdx.x * 4) = h;
}

// ---------------------------------------------------------------------------
// Kernel 3: out[M,HID] = aggH[M,CTX] @ WH[CTX,HID]   (f16 in, f32 accum, WMMA)
// A tile (128x32 halfs) staged by the Tensor Data Mover into LDS with HW
// padding (16-DWORD interval, 4-DWORD pad -> 80B row stride). B tile staged +
// transposed by the 8 waves. 8 v_wmma_f32_16x16x32_f16 per wave per K-step.
// ---------------------------------------------------------------------------
__global__ __launch_bounds__(256) void gemm_kernel(const _Float16* __restrict__ A,
                                                   const _Float16* __restrict__ Bw,
                                                   float* __restrict__ out) {
    extern __shared__ _Float16 smem[];
    _Float16* AsT = smem;             // LDS byte offset 0      (TDM destination)
    _Float16* Bt  = smem + BM * AS;   // LDS byte offset 10240  Bt[col][k]

    const int tid   = threadIdx.x;
    const int lane  = tid & 31;
    const int w     = tid >> 5;
    const int wr    = w >> 2;          // 0..1
    const int wc    = w & 3;           // 0..3
    const int lrow  = lane & 15;
    const int lhalf = lane >> 4;
    const int ko    = lhalf * 8;       // K base offset of this lane's fragment half
    const int m0    = blockIdx.y * BM;
    const int n0    = blockIdx.x * BN;

    const unsigned long long abase =
        (unsigned long long)A + ((unsigned long long)m0 * CTXD) * 2ull;

    v8f acc[4][2] = {};

    for (int kt = 0; kt < CTXD; kt += BK) {
        // --- wave 0: issue TDM load of the A tile (128 rows x 32 halfs) ---
        if (tid < 32) {
            unsigned long long ga = abase + (unsigned long long)kt * 2ull;
            u32x4 g0;
            g0[0] = 1u;                                   // count=1, user mode
            g0[1] = 0u;                                   // lds_addr = 0 (AsT)
            g0[2] = (unsigned)(ga & 0xFFFFFFFFu);         // global_addr[31:0]
            g0[3] = (unsigned)((ga >> 32) & 0x01FFFFFFu)  // global_addr[56:32]
                    | (2u << 30);                         // type = 2 ("image")
            i32x8 g1;
            g1[0] = (int)((1u << 16)    // data_size = 2 bytes
                    | (1u << 20)        // pad_enable
                    | (3u << 22)        // pad_interval: 16 DWORDs (one 32-half row)
                    | (3u << 25));      // pad_amount: 4 DWORDs -> 80B row stride
            g1[1] = (int)(((unsigned)CTXD & 0xFFFFu) << 16);   // tensor_dim0 = 512
            g1[2] = (int)(((unsigned)MTOT & 0xFFFFu) << 16);   // tensor_dim1 = 32768 (lo16)
            g1[3] = (int)(32u << 16);                          // tile_dim0 = 32 halfs
            g1[4] = 128;                                       // tile_dim1 = 128 rows
            g1[5] = CTXD;                                      // tensor_dim0_stride = 512
            g1[6] = 0;
            g1[7] = 0;
            i32x4 z4 = {0, 0, 0, 0};
#if __clang_major__ >= 23
            i32x8 z8 = {0, 0, 0, 0, 0, 0, 0, 0};
            __builtin_amdgcn_tensor_load_to_lds(g0, g1, z4, z4, z8, 0);
#else
            __builtin_amdgcn_tensor_load_to_lds(g0, g1, z4, z4, 0);
#endif
        }

        // --- all waves: stage B tile (32 x 128 halfs) transposed into Bt ---
#pragma unroll
        for (int c = 0; c < 2; ++c) {
            int chunk = tid + c * 256;            // 0..511
            int k    = chunk >> 4;                // 0..31
            int coff = (chunk & 15) * 8;          // 0..120
            v8h v = *(const v8h*)(Bw + (size_t)(kt + k) * HIDD + n0 + coff);
#pragma unroll
            for (int j = 0; j < 8; ++j)
                Bt[(coff + j) * BS + k] = v[j];
        }

        if (tid < 32)
            __builtin_amdgcn_s_wait_tensorcnt(0);
        __syncthreads();

        // --- build fragments (two aligned b128 LDS reads each) ---
        v16h af[4], bf[2];
#pragma unroll
        for (int i = 0; i < 4; ++i) {
            int r = wr * 64 + i * 16 + lrow;
            v8h a0 = *(const v8h*)(&AsT[r * AS + ko]);
            v8h a1 = *(const v8h*)(&AsT[r * AS + ko + 16]);
            af[i] = __builtin_shufflevector(a0, a1, 0, 1, 2, 3, 4, 5, 6, 7,
                                                    8, 9, 10, 11, 12, 13, 14, 15);
        }
#pragma unroll
        for (int j = 0; j < 2; ++j) {
            int cidx = wc * 32 + j * 16 + lrow;
            v8h b0 = *(const v8h*)(&Bt[cidx * BS + ko]);
            v8h b1 = *(const v8h*)(&Bt[cidx * BS + ko + 16]);
            bf[j] = __builtin_shufflevector(b0, b1, 0, 1, 2, 3, 4, 5, 6, 7,
                                                    8, 9, 10, 11, 12, 13, 14, 15);
        }

        // --- 8 WMMAs per wave per K-step ---
#pragma unroll
        for (int i = 0; i < 4; ++i)
#pragma unroll
            for (int j = 0; j < 2; ++j)
                acc[i][j] = __builtin_amdgcn_wmma_f32_16x16x32_f16(
                    /*neg_a=*/false, af[i], /*neg_b=*/false, bf[j],
                    /*c_mod=*/(short)0, acc[i][j],
                    /*reuse_a=*/false, /*reuse_b=*/false);
        __syncthreads();
    }

    // --- writeback: C layout: VGPR e -> row e (lanes 0-15) / e+8 (lanes 16-31) ---
#pragma unroll
    for (int i = 0; i < 4; ++i) {
#pragma unroll
        for (int j = 0; j < 2; ++j) {
            int gr0 = m0 + wr * 64 + i * 16 + lhalf * 8;
            int gc  = n0 + wc * 32 + j * 16 + lrow;
#pragma unroll
            for (int e = 0; e < 8; ++e)
                out[(size_t)(gr0 + e) * HIDD + gc] = acc[i][j][e];
        }
    }
}

// ---------------------------------------------------------------------------
extern "C" void kernel_launch(void* const* d_in, const int* in_sizes, int n_in,
                              void* d_out, int out_size, void* d_ws, size_t ws_size,
                              hipStream_t stream) {
    const float* x   = (const float*)d_in[0];   // (B,H,W,CTX) f32
    const float* W   = (const float*)d_in[1];   // (CTX,HID)  f32
    const int*   nn  = (const int*)d_in[2];     // (N,K)      i32
    const float* sim = (const float*)d_in[3];   // (N,K)      f32
    float* out = (float*)d_out;                 // (B,H,W,HID) f32

    _Float16* aggH = (_Float16*)d_ws;                                     // 33.5 MB
    _Float16* WH   = (_Float16*)((char*)d_ws +
                                 (size_t)MTOT * CTXD * sizeof(_Float16)); // +2 MB

    convw_kernel<<<dim3((CTXD * HIDD) / 1024), 256, 0, stream>>>(W, WH);
    agg_kernel<<<dim3(NCELL, BATCH), 128, 0, stream>>>(x, nn, sim, aggH);
    gemm_kernel<<<dim3(HIDD / BN, MTOT / BM), 256, LDS_BYTES, stream>>>(aggH, WH, out);
}
